// Attention_10282151706748
// MI455X (gfx1250) — compile-verified, matrix operand order checked by reading, not run on hardware
//
#include <hip/hip_runtime.h>
#include <math.h>

typedef __attribute__((ext_vector_type(16))) _Float16 v16h;
typedef __attribute__((ext_vector_type(8)))  float    v8f;

constexpr int B_  = 32;
constexpr int L_  = 8192;
constexpr int IN_ = 512;
constexpr int HID_ = 512;
constexpr int D3_ = 128;

constexpr int KSTEPS = HID_ / 32;      // 16 K-steps of 32
constexpr int NTILES = D3_ / 16;       // 8 N-tiles of 16
constexpr int TILES  = B_ * (L_ / 16); // 16384 row-tiles of 16 L-rows
constexpr int WAVES_PER_BLOCK = 8;
constexpr int TILES_PER_WAVE  = 8;
constexpr int NBLOCKS = TILES / (WAVES_PER_BLOCK * TILES_PER_WAVE); // 256
constexpr int WH_ELEMS = KSTEPS * NTILES * 32 * 16;                 // 65536 = all of Wh

// ---------------- Kernel 1: s[b,d] = (inputs @ Wi^T)[b,d] + bi[d] + bh[d] ----------------
__global__ __launch_bounds__(128) void qk_kernel(const float* __restrict__ inputs,
                                                 const float* __restrict__ Wi,
                                                 const float* __restrict__ bi,
                                                 const float* __restrict__ bh,
                                                 float* __restrict__ s) {
    int b = blockIdx.x;      // 32
    int d = threadIdx.x;     // 128
    const float* in = inputs + (size_t)b * IN_;
    const float* w  = Wi     + (size_t)d * IN_;
    float acc = 0.f;
    for (int h = 0; h < IN_; h += 4) {
        float4 a  = *(const float4*)(in + h);
        float4 ww = *(const float4*)(w + h);
        acc += a.x * ww.x + a.y * ww.y + a.z * ww.z + a.w * ww.w;
    }
    s[b * D3_ + d] = acc + bi[d] + bh[d];
}

// ---------------- Kernel 2: fused context@Wh^T + tanh(q+k) dot V, softmax(=1) ----------------
__global__ __launch_bounds__(256) void attn_kernel(const float* __restrict__ ctx,
                                                   const float* __restrict__ Wh,
                                                   const float* __restrict__ s,
                                                   const float* __restrict__ V,
                                                   float* __restrict__ out_attrow,
                                                   float* __restrict__ out_att) {
    // Wh staged once per block, pre-swizzled into WMMA-B fragment order:
    // fragment (kk, n): lane holds column N = n*16 + (lane&15), 16 f16 K-values
    // at h = kk*32 + ((lane>=16)?8:0) + (e/8)*16 + (e%8). One v16h = 32B contiguous.
    __shared__ __align__(32) _Float16 whlds[WH_ELEMS];   // 128 KB of the 320 KB WGP LDS

    int tid = threadIdx.x;
    for (int i = tid; i < WH_ELEMS; i += 256) {
        int e    = i & 15;
        int lane = (i >> 4) & 31;
        int nk   = i >> 9;            // kk*8 + n
        int n    = nk & 7;
        int kk   = nk >> 3;
        int d    = n * 16 + (lane & 15);
        int h    = kk * 32 + ((lane >> 4) << 3) + ((e >> 3) << 4) + (e & 7);
        whlds[i] = (_Float16)Wh[(size_t)d * HID_ + h];
    }
    __syncthreads();

    int wave = tid >> 5;
    int lane = tid & 31;
    const v16h* bfr = (const v16h*)whlds;
    int tile0 = (blockIdx.x * WAVES_PER_BLOCK + wave) * TILES_PER_WAVE;

    for (int t = 0; t < TILES_PER_WAVE; ++t) {
        int tile = tile0 + t;
        int b  = tile / (L_ / 16);
        int l0 = (tile % (L_ / 16)) * 16;

        // A fragment source: this lane's context row, with half-wave K split
        const float* arow = ctx + ((size_t)b * L_ + l0 + (lane & 15)) * (size_t)HID_;
        int koff = (lane >> 4) << 3;   // lanes 16-31 take K+8

        v8f acc[NTILES];
        v8f zero = 0.0f;
#pragma unroll
        for (int n = 0; n < NTILES; ++n) acc[n] = zero;

        for (int kk = 0; kk < KSTEPS; ++kk) {
            int kb = kk * 32 + koff;
            float4 f0 = *(const float4*)(arow + kb);
            float4 f1 = *(const float4*)(arow + kb + 4);
            float4 f2 = *(const float4*)(arow + kb + 16);
            float4 f3 = *(const float4*)(arow + kb + 20);
            v16h a;
            a[0]  = (_Float16)f0.x; a[1]  = (_Float16)f0.y;
            a[2]  = (_Float16)f0.z; a[3]  = (_Float16)f0.w;
            a[4]  = (_Float16)f1.x; a[5]  = (_Float16)f1.y;
            a[6]  = (_Float16)f1.z; a[7]  = (_Float16)f1.w;
            a[8]  = (_Float16)f2.x; a[9]  = (_Float16)f2.y;
            a[10] = (_Float16)f2.z; a[11] = (_Float16)f2.w;
            a[12] = (_Float16)f3.x; a[13] = (_Float16)f3.y;
            a[14] = (_Float16)f3.z; a[15] = (_Float16)f3.w;
#pragma unroll
            for (int n = 0; n < NTILES; ++n) {
                v16h bf = bfr[(kk * NTILES + n) * 32 + lane];
                acc[n] = __builtin_amdgcn_wmma_f32_16x16x32_f16(
                    false, a, false, bf, (short)0, acc[n], false, false);
            }
        }

        // Fused epilogue: tanh(acc + s[b,d]) * V[d], reduce over d.
        // acc[n] VGPR v: lanes 0-15 -> row v, lanes 16-31 -> row v+8; N = lane&15.
        float partial[8];
#pragma unroll
        for (int v = 0; v < 8; ++v) partial[v] = 0.f;
        const float* srow = s + b * D3_;
#pragma unroll
        for (int n = 0; n < NTILES; ++n) {
            float sv = srow[n * 16 + (lane & 15)];
            float vv = V[n * 16 + (lane & 15)];
#pragma unroll
            for (int v = 0; v < 8; ++v)
                partial[v] += tanhf(acc[n][v] + sv) * vv;
        }
        // reduce across the 16-lane N groups (xor masks < 16 stay in-group)
#pragma unroll
        for (int off = 1; off < 16; off <<= 1) {
#pragma unroll
            for (int v = 0; v < 8; ++v)
                partial[v] += __shfl_xor(partial[v], off, 32);
        }

        if ((lane & 15) == 0) {
            int r0 = (lane >> 4) << 3;  // lane 0 -> rows 0..7, lane 16 -> rows 8..15
            float* o1 = out_attrow + (size_t)b * L_ + l0 + r0;
            float* o2 = out_att    + (size_t)b * L_ + l0 + r0;
#pragma unroll
            for (int v = 0; v < 8; ++v) {
                o1[v] = partial[v];
                o2[v] = 1.0f;   // softmax over singleton dim == 1 (faithful quirk)
            }
        }
    }
}

extern "C" void kernel_launch(void* const* d_in, const int* in_sizes, int n_in,
                              void* d_out, int out_size, void* d_ws, size_t ws_size,
                              hipStream_t stream) {
    const float* inputs  = (const float*)d_in[0];
    const float* context = (const float*)d_in[1];
    const float* Wi      = (const float*)d_in[2];
    const float* bi      = (const float*)d_in[3];
    const float* Wh      = (const float*)d_in[4];
    const float* bh      = (const float*)d_in[5];
    const float* V       = (const float*)d_in[6];
    float* out = (float*)d_out;
    float* s   = (float*)d_ws;   // 32*128 floats = 16 KB scratch

    qk_kernel<<<B_, 128, 0, stream>>>(inputs, Wi, bi, bh, s);
    attn_kernel<<<NBLOCKS, 256, 0, stream>>>(context, Wh, s, V,
                                             out,                      // att_row (B,1,L)
                                             out + (size_t)B_ * L_);   // att     (B,L)
}
